// TreeGradient_36043365548588
// MI455X (gfx1250) — compile-verified, matrix operand order checked by reading, not run on hardware
//
#include <hip/hip_runtime.h>
#include <hip/hip_bf16.h>

// ---------------- problem constants ----------------
#define NN      192
#define MMd     256
#define DD      1024
#define KCONV   65
#define OUTW    192
#define NT      16
#define NTILES  (NN / NT)          // 12
#define CHUNK   64
#define NCHUNK  (DD / CHUNK)       // 16
#define WG      512                // 16 waves, one wave per 16-col block

#define RPAD    (MMd + 4)          // f32 LDS row stride
#define APAD    (MMd + 8)          // bf16 LDS row stride (264 ushorts = 528 B, 16B aligned)

typedef __bf16 v16bf __attribute__((ext_vector_type(16)));
typedef float  v8f   __attribute__((ext_vector_type(8)));
typedef unsigned int uint32;
typedef int async_v4i __attribute__((__vector_size__(4 * sizeof(int))));

#if defined(__AMDGCN__) && __has_builtin(__builtin_amdgcn_global_load_async_to_lds_b128) && __has_builtin(__builtin_amdgcn_s_wait_asynccnt)
#define USE_ASYNC_LDS 1
#else
#define USE_ASYNC_LDS 0
#endif

// ---------------- helpers ----------------
__device__ __forceinline__ uint32 pack_bf16_2(float a, float b) {
    uint32 ua = __builtin_bit_cast(uint32, a);
    uint32 ub = __builtin_bit_cast(uint32, b);
    ua += 0x7FFFu + ((ua >> 16) & 1u);          // RNE to bf16
    ub += 0x7FFFu + ((ub >> 16) & 1u);
    return (ua >> 16) | (ub & 0xFFFF0000u);
}
__device__ __forceinline__ unsigned short bf16_1(float a) {
    uint32 u = __builtin_bit_cast(uint32, a);
    u += 0x7FFFu + ((u >> 16) & 1u);
    return (unsigned short)(u >> 16);
}
__device__ __forceinline__ float bf16f(unsigned short h) {
    uint32 u = ((uint32)h) << 16;
    return __builtin_bit_cast(float, u);
}
__device__ __forceinline__ float sigmoidf_(float x) {
    return 1.0f / (1.0f + __expf(-x));
}

union FragU { uint32 u[8]; v16bf v; };

// pre-packed bf16 B fragment (32 lanes x 8 dwords) from global
__device__ __forceinline__ v16bf load_frag_g(const uint32* fragbase, int lane) {
    FragU r;
    const uint32* p = fragbase + lane * 8;
#pragma unroll
    for (int i = 0; i < 8; ++i) r.u[i] = p[i];
    return r.v;
}

// A fragment (16x32 bf16) from row-major bf16 LDS tile.
// Lane L (M = L&15): dword v<4 holds K={2v,2v+1}, v>=4 holds K={2v+8,2v+9};
// lanes 16..31 add K offset 8.  -> two contiguous 16B groups per kb => ds_load_b128 x2
__device__ __forceinline__ v16bf afrag_bf16(const unsigned short* rowp, int halfk, int kb) {
    FragU r;
#pragma unroll
    for (int v = 0; v < 8; ++v) {
        int k = kb * 32 + ((v < 4) ? 2 * v : 8 + 2 * v) + halfk * 8;
        r.u[v] = *(const uint32*)(rowp + k);
    }
    return r.v;
}

// A fragment from row-major f32 LDS (only used in k_combine, off hot loop)
__device__ __forceinline__ v16bf afrag_f32(const float* rowp, int halfk, int kb) {
    FragU r;
#pragma unroll
    for (int v = 0; v < 8; ++v) {
        int k = kb * 32 + ((v < 4) ? 2 * v : 8 + 2 * v) + halfk * 8;
        float2 f = *(const float2*)(rowp + k);
        r.u[v] = pack_bf16_2(f.x, f.y);
    }
    return r.v;
}

// ---------------- prep kernels ----------------
__global__ void k_split(const float* __restrict__ W,
                        float* __restrict__ W0f, float* __restrict__ Pf) {
    int m = blockIdx.x, k = threadIdx.x;
    size_t s = ((size_t)k * MMd + m) * 2;
    W0f[m * MMd + k] = W[s + 0];           // W0^T
    Pf[m * MMd + k]  = W[s + 1];           // P = W1^T
}

__global__ void k_mm256(const float* __restrict__ A, const float* __restrict__ B,
                        float* __restrict__ C) {
    int row = blockIdx.x, col = threadIdx.x;
    float s = 0.f;
    for (int k = 0; k < MMd; ++k) s = fmaf(A[row * MMd + k], B[k * MMd + col], s);
    C[row * MMd + col] = s;
}

// pack row-major f32 B matrix into WMMA bf16 B fragments (frag = (kb*16+nb), 32 lanes x 8 dw)
__global__ void k_packB(const float* __restrict__ src, uint32* __restrict__ dst) {
    int tid = threadIdx.x;
    int wave = tid >> 5, lane = tid & 31;
    int fid = blockIdx.x * 8 + wave;
    int kb = fid >> 4, nb = fid & 15;
    int lcol = lane & 15, halfk = lane >> 4;
    int col = nb * 16 + lcol;
    uint32* d = dst + fid * 256 + lane * 8;
#pragma unroll
    for (int v = 0; v < 8; ++v) {
        int k = kb * 32 + ((v < 4) ? 2 * v : 8 + 2 * v) + halfk * 8;
        d[v] = pack_bf16_2(src[k * MMd + col], src[(k + 1) * MMd + col]);
    }
}

// ---------------- main fused chunk kernel ----------------
// acc <- acc @ P + (rows_j @ W0^T + b + sigmoid(rows_j)),  j descending in chunk;
// j == 1023 seeds acc <- h0.  Double-buffered bf16 rows tile, pipelined global loads.
__global__ __launch_bounds__(WG) void k_chunk(const float* __restrict__ NATree,
                                              const uint32* __restrict__ W0pack,
                                              const uint32* __restrict__ Ppack,
                                              const float* __restrict__ bvec,
                                              float* __restrict__ Tbuf) {
    __shared__ unsigned short rowsA[2][NT][APAD];
    __shared__ unsigned short accs[NT][APAD];

    const int t = blockIdx.x, c = blockIdx.y;
    const int tid = threadIdx.x;
    const int wave = tid >> 5, lane = tid & 31;
    const int nb = wave;
    const int lcol = lane & 15, halfk = lane >> 4;
    const int col = nb * 16 + lcol;

    // B operands register-resident for all 64 steps
    v16bf w0f[8], pf[8];
#pragma unroll
    for (int kb = 0; kb < 8; ++kb) {
        w0f[kb] = load_frag_g(W0pack + (kb * 16 + nb) * 256, lane);
        pf[kb]  = load_frag_g(Ppack  + (kb * 16 + nb) * 256, lane);
    }
    const float bcol = bvec[col];

    const int ldrow = tid >> 5;             // 32 threads/row, 8 elems each
    const int ldc0  = (tid & 31) * 8;
    const float* srcbase = NATree + ((size_t)(t * NT + ldrow) * DD) * MMd + ldc0;

    // zero acc, preload first rows tile (j = c*64+63) as bf16
    for (int i = tid; i < NT * APAD; i += WG) (&accs[0][0])[i] = 0;
    {
        const float* s0 = srcbase + (size_t)(c * CHUNK + CHUNK - 1) * MMd;
        float4 ra = ((const float4*)s0)[0];
        float4 rb = ((const float4*)s0)[1];
        uint4 pk = {pack_bf16_2(ra.x, ra.y), pack_bf16_2(ra.z, ra.w),
                    pack_bf16_2(rb.x, rb.y), pack_bf16_2(rb.z, rb.w)};
        *(uint4*)(&rowsA[0][ldrow][ldc0]) = pk;
    }
    __syncthreads();

    v8f accv = {0, 0, 0, 0, 0, 0, 0, 0};
    int cur = 0;

    for (int r = CHUNK - 1; r >= 0; --r) {
        const int j = c * CHUNK + r;

        // issue next step's global loads early (consumed after the WMMA block)
        float4 ra, rb;
        if (r > 0) {
            const float* sn = srcbase + (size_t)(j - 1) * MMd;
            ra = ((const float4*)sn)[0];
            rb = ((const float4*)sn)[1];
        }

        const unsigned short* rowtile = &rowsA[cur][0][0];
        v8f res;
        if (j == 1023) {                    // seed with h0 = NATree[:, -1, :]
#pragma unroll
            for (int v = 0; v < 8; ++v)
                res[v] = bf16f(rowtile[(v + halfk * 8) * APAD + col]);
        } else {
            v8f cu = {0, 0, 0, 0, 0, 0, 0, 0};
#pragma unroll
            for (int kb = 0; kb < 8; ++kb) {   // acc @ P
                v16bf a = afrag_bf16(&accs[lcol][0], halfk, kb);
                cu = __builtin_amdgcn_wmma_f32_16x16x32_bf16(
                         false, a, false, pf[kb], (short)0, cu, false, false);
            }
#pragma unroll
            for (int kb = 0; kb < 8; ++kb) {   // rows @ W0^T
                v16bf a = afrag_bf16(rowtile + lcol * APAD, halfk, kb);
                cu = __builtin_amdgcn_wmma_f32_16x16x32_bf16(
                         false, a, false, w0f[kb], (short)0, cu, false, false);
            }
#pragma unroll
            for (int v = 0; v < 8; ++v) {
                float rv = bf16f(rowtile[(v + halfk * 8) * APAD + col]);
                res[v] = cu[v] + bcol + sigmoidf_(rv);
            }
        }

        // stage next rows tile (bf16) into the other buffer
        if (r > 0) {
            uint4 pk = {pack_bf16_2(ra.x, ra.y), pack_bf16_2(ra.z, ra.w),
                        pack_bf16_2(rb.x, rb.y), pack_bf16_2(rb.z, rb.w)};
            *(uint4*)(&rowsA[cur ^ 1][ldrow][ldc0]) = pk;
        }
        __syncthreads();     // acc/rows reads done; next rows stored

#pragma unroll
        for (int v = 0; v < 8; ++v)
            accs[v + halfk * 8][col] = bf16_1(res[v]);
        __syncthreads();     // acc published

        accv = res;
        cur ^= 1;
    }

    // T_c tile -> scratch (f32)
#pragma unroll
    for (int v = 0; v < 8; ++v)
        Tbuf[((size_t)(c * NTILES + t) * NT + v + halfk * 8) * MMd + col] = accv[v];
}

// ---------------- combine + epilogue ----------------
__global__ __launch_bounds__(WG) void k_combine(const float* __restrict__ Tbuf,
                                                const uint32* __restrict__ Gpack,
                                                const float* __restrict__ x,
                                                const float* __restrict__ W2,
                                                const float* __restrict__ b2,
                                                float* __restrict__ out) {
    __shared__ float tb[NT][RPAD];
    __shared__ float w2s[KCONV];

    const int t = blockIdx.x;
    const int tid = threadIdx.x;
    const int wave = tid >> 5, lane = tid & 31;
    const int nb = wave;
    const int lcol = lane & 15, halfk = lane >> 4;
    const int col = nb * 16 + lcol;
    const int ldrow = tid >> 5, ldc0 = (tid & 31) * 8;

    if (tid < KCONV) w2s[tid] = W2[tid];

    v8f acc = {0, 0, 0, 0, 0, 0, 0, 0};
    for (int c = 0; c < NCHUNK; ++c) {
        __syncthreads();
        const float* src = Tbuf + ((size_t)(c * NTILES + t) * NT + ldrow) * MMd + ldc0;
        float* dst = &tb[ldrow][ldc0];
#if USE_ASYNC_LDS
        __builtin_amdgcn_global_load_async_to_lds_b128(
            (async_v4i*)const_cast<float*>(src), (async_v4i*)dst, 0, 0);
        __builtin_amdgcn_global_load_async_to_lds_b128(
            (async_v4i*)const_cast<float*>(src + 4), (async_v4i*)(dst + 4), 0, 0);
        __builtin_amdgcn_s_wait_asynccnt(0);
#else
        *(float4*)(dst)     = ((const float4*)src)[0];
        *(float4*)(dst + 4) = ((const float4*)src)[1];
#endif
        __syncthreads();
        if (c == 0) {                       // G_0 = I
#pragma unroll
            for (int v = 0; v < 8; ++v) acc[v] += tb[v + halfk * 8][col];
        } else {
            const uint32* gp = Gpack + (size_t)(c - 1) * 128 * 256;
#pragma unroll
            for (int kb = 0; kb < 8; ++kb) {
                v16bf a = afrag_f32(&tb[lcol][0], halfk, kb);
                v16bf g = load_frag_g(gp + (kb * 16 + nb) * 256, lane);
                acc = __builtin_amdgcn_wmma_f32_16x16x32_bf16(
                          false, a, false, g, (short)0, acc, false, false);
            }
        }
    }

    __syncthreads();
#pragma unroll
    for (int v = 0; v < 8; ++v) {           // relu(h_fin + x)
        int n = t * NT + v + halfk * 8;
        float rv = acc[v] + x[(size_t)n * MMd + col];
        tb[v + halfk * 8][col] = rv > 0.f ? rv : 0.f;
    }
    __syncthreads();

    const float bias2 = b2[0];
    for (int o = tid; o < NT * OUTW; o += WG) {
        int rr = o / OUTW, ii = o % OUTW;
        float s = bias2;
#pragma unroll 5
        for (int kk = 0; kk < KCONV; ++kk) s = fmaf(tb[rr][ii + kk], w2s[kk], s);
        out[(size_t)(t * NT + rr) * OUTW + ii] = sigmoidf_(s);
    }
}

// ---------------- host launcher ----------------
extern "C" void kernel_launch(void* const* d_in, const int* in_sizes, int n_in,
                              void* d_out, int out_size, void* d_ws, size_t ws_size,
                              hipStream_t stream) {
    const float* NATree = (const float*)d_in[0];
    const float* W      = (const float*)d_in[1];
    const float* bvec   = (const float*)d_in[2];
    const float* x      = (const float*)d_in[3];
    const float* W2     = (const float*)d_in[4];
    const float* b2     = (const float*)d_in[5];
    float* out = (float*)d_out;

    float* ws = (float*)d_ws;
    const size_t MAT = (size_t)MMd * MMd;
    float* W0f  = ws;
    float* Pf   = W0f + MAT;
    float* tmpA = Pf + MAT;
    float* tmpB = tmpA + MAT;
    float* G    = tmpB + MAT;                         // 15 x MAT : P^{64c}
    float* Tbuf = G + 15 * MAT;
    uint32* W0pack = (uint32*)(Tbuf + (size_t)NCHUNK * NTILES * NT * MMd);
    uint32* Ppack  = W0pack + 128 * 256;
    uint32* Gpack  = Ppack  + 128 * 256;

    k_split<<<MMd, MMd, 0, stream>>>(W, W0f, Pf);

    k_mm256<<<MMd, MMd, 0, stream>>>(Pf,   Pf,   tmpA);   // P^2
    k_mm256<<<MMd, MMd, 0, stream>>>(tmpA, tmpA, tmpB);   // P^4
    k_mm256<<<MMd, MMd, 0, stream>>>(tmpB, tmpB, tmpA);   // P^8
    k_mm256<<<MMd, MMd, 0, stream>>>(tmpA, tmpA, tmpB);   // P^16
    k_mm256<<<MMd, MMd, 0, stream>>>(tmpB, tmpB, tmpA);   // P^32
    k_mm256<<<MMd, MMd, 0, stream>>>(tmpA, tmpA, G);      // P^64 = G_1
    for (int c = 2; c <= 15; ++c)
        k_mm256<<<MMd, MMd, 0, stream>>>(G + (size_t)(c - 2) * MAT, G,
                                         G + (size_t)(c - 1) * MAT);

    k_packB<<<16, 256, 0, stream>>>(W0f, W0pack);
    k_packB<<<16, 256, 0, stream>>>(Pf,  Ppack);
    for (int c = 1; c <= 15; ++c)
        k_packB<<<16, 256, 0, stream>>>(G + (size_t)(c - 1) * MAT,
                                        Gpack + (size_t)(c - 1) * 128 * 256);

    k_chunk<<<dim3(NTILES, NCHUNK), WG, 0, stream>>>(NATree, W0pack, Ppack, bvec, Tbuf);
    k_combine<<<NTILES, WG, 0, stream>>>(Tbuf, Gpack, x, W2, b2, out);
}
